// DUSTED_85899345920188
// MI455X (gfx1250) — compile-verified, matrix operand order checked by reading, not run on hardware
//
#include <hip/hip_runtime.h>
#include <hip/hip_bf16.h>
#include <math.h>

#define N_NODES 50000
#define N_EDGES 800000
#define C_IN    1024
#define C_HID   512
#define C_OUT   32
#define C_RED   64

typedef __bf16 bf16_t;
typedef __attribute__((ext_vector_type(16))) __bf16 v16bf;
typedef __attribute__((ext_vector_type(8)))  __bf16 v8bf;
typedef __attribute__((ext_vector_type(4)))  __bf16 v4bf;
typedef __attribute__((ext_vector_type(8)))  float  v8f;
typedef __attribute__((ext_vector_type(4)))  float  v4fe;

// ---------------------------------------------------------------- utilities

__device__ __forceinline__ float atomicMaxF(float* addr, float v) {
  if (v >= 0.f) {
    return __int_as_float(atomicMax((int*)addr, __float_as_int(v)));
  } else {
    return __uint_as_float(atomicMin((unsigned int*)addr, __float_as_uint(v)));
  }
}

__global__ void fill_f32(float* __restrict__ p, float v, size_t n) {
  size_t i  = (size_t)blockIdx.x * blockDim.x + threadIdx.x;
  size_t st = (size_t)gridDim.x * blockDim.x;
  for (; i < n; i += st) p[i] = v;
}

// ------------------------------------------------ GCA: column mean/max pool

__global__ void col_reduce(const float* __restrict__ X, float* __restrict__ sum,
                           float* __restrict__ mx, int Nn, int Cc, int rowsPerBlk) {
  int c = blockIdx.x * blockDim.x + threadIdx.x;
  if (c >= Cc) return;
  int r0 = blockIdx.y * rowsPerBlk;
  int r1 = r0 + rowsPerBlk; if (r1 > Nn) r1 = Nn;
  float s = 0.f, m = -__builtin_inff();
  for (int r = r0; r < r1; ++r) {
    float v = X[(size_t)r * Cc + c];
    s += v; m = fmaxf(m, v);
  }
  atomicAdd(&sum[c], s);
  atomicMaxF(&mx[c], m);
}

// shared-MLP channel gate: scale[c] = 1 + 1.5*sigmoid(att[c])
__global__ void gca_mlp(const float* __restrict__ sum, const float* __restrict__ mx,
                        const float* __restrict__ gw1, const float* __restrict__ gb1,
                        const float* __restrict__ gw2, const float* __restrict__ gb2,
                        float* __restrict__ scaleOut, int Nn) {
  __shared__ float rh[2 * C_RED];
  int t = threadIdx.x;
  if (t < C_RED) {
    float sa = gb1[t], sm = gb1[t];
    for (int c = 0; c < C_IN; ++c) {
      float w = gw1[(size_t)c * C_RED + t];
      sa += (sum[c] / (float)Nn) * w;
      sm += mx[c] * w;
    }
    rh[t]         = fmaxf(sa, 0.f);
    rh[C_RED + t] = fmaxf(sm, 0.f);
  }
  __syncthreads();
  float att = 2.f * gb2[t];
  for (int j = 0; j < C_RED; ++j)
    att += (rh[j] + rh[C_RED + j]) * gw2[(size_t)j * C_IN + t];
  scaleOut[t] = 1.f + 1.5f / (1.f + __expf(-att));
}

// ------------------------------------------------------- WMMA bf16 GEMM
// C[M,BN*gridY] = (A * gate?) @ B.  Block tile 64 x (32*COLT), BK=32.
// 8 wave32 waves: 4 row-tiles x 2 col-groups; COLT WMMAs per wave per k-step
// sharing one A fragment.  LDS rows padded to 80B so every fragment is two
// ds_load_b128 (CDNA5 16-bit operand layout: lane half h holds K 8h..8h+7, 16+8h..16+8h+7).

template <int COLT, bool TRANSB, bool SCALED>
__global__ __launch_bounds__(256) void gemm_wmma(
    const float* __restrict__ A, int lda, const float* __restrict__ colscale,
    const float* __restrict__ B, int ldb, float* __restrict__ C, int ldc,
    int M, int K) {
  constexpr int BN = 32 * COLT;
  __shared__ bf16_t As[64][40];   // row stride 80B (16B aligned, bank-spread)
  __shared__ bf16_t Bs[BN][40];   // stored transposed: Bs[n][k]
  const int tid   = threadIdx.x;
  const int lane  = tid & 31;
  const int wave  = tid >> 5;
  const int mBase = blockIdx.x * 64;
  const int nBase = blockIdx.y * BN;
  const int m0    = (wave & 3) * 16;
  const int n0    = (wave >> 2) * 16 * COLT;

  v8f zero = {};
  v8f acc[COLT];
#pragma unroll
  for (int t = 0; t < COLT; ++t) acc[t] = zero;

  for (int k0 = 0; k0 < K; k0 += 32) {
    // stage A: 64x32 fp32 as float4 loads, fused gate, pack to bf16x4
#pragma unroll
    for (int i = 0; i < 2; ++i) {
      int li = i * 256 + tid;
      int r = li >> 3, c4 = (li & 7) * 4;
      int gr = mBase + r;
      v4fe av = {};
      if (gr < M) av = *(const v4fe*)(A + (size_t)gr * lda + (k0 + c4));
      if (SCALED) av *= *(const v4fe*)(colscale + (k0 + c4));
      *(v4bf*)&As[r][c4] = __builtin_convertvector(av, v4bf);
    }
    // stage B transposed into Bs[n][k]
#pragma unroll
    for (int i = 0; i < BN / 8; ++i) {
      int li = i * 256 + tid;
      int n, k;
      float v;
      if (TRANSB) {
        k = li & 31; n = li >> 5;
        v = B[(size_t)(nBase + n) * ldb + (k0 + k)];
      } else {
        n = li % BN; k = li / BN;
        v = B[(size_t)(k0 + k) * ldb + (nBase + n)];
      }
      Bs[n][k] = (bf16_t)v;
    }
    __syncthreads();

    // fragments: two b128 LDS loads each (K kb..kb+7 | kb+16..kb+23)
    const int mi = m0 + (lane & 15);
    const int kb = (lane >> 4) << 3;
    v8bf alo = *(const v8bf*)&As[mi][kb];
    v8bf ahi = *(const v8bf*)&As[mi][kb + 16];
    v16bf a = __builtin_shufflevector(alo, ahi, 0, 1, 2, 3, 4, 5, 6, 7,
                                      8, 9, 10, 11, 12, 13, 14, 15);
#pragma unroll
    for (int t = 0; t < COLT; ++t) {
      int ni = n0 + t * 16 + (lane & 15);
      v8bf blo = *(const v8bf*)&Bs[ni][kb];
      v8bf bhi = *(const v8bf*)&Bs[ni][kb + 16];
      v16bf b = __builtin_shufflevector(blo, bhi, 0, 1, 2, 3, 4, 5, 6, 7,
                                        8, 9, 10, 11, 12, 13, 14, 15);
      acc[t] = __builtin_amdgcn_wmma_f32_16x16x32_bf16(
          false, a, false, b, (short)0, acc[t], false, false);
    }
    __syncthreads();
  }

  // C/D layout: lane<16 -> rows m0+v, lane>=16 -> rows m0+8+v; col = lane&15
  const int mrow = mBase + m0 + ((lane >> 4) & 1) * 8;
#pragma unroll
  for (int t = 0; t < COLT; ++t) {
    int n = nBase + n0 + t * 16 + (lane & 15);
#pragma unroll
    for (int vv = 0; vv < 8; ++vv) {
      int m = mrow + vv;
      if (m < M) C[(size_t)m * ldc + n] = acc[t][vv];
    }
  }
}

// -------------------------------------------- per-node attention score dots

__global__ void node_dots(const float* __restrict__ H, const float* __restrict__ vs,
                          const float* __restrict__ vd, float* __restrict__ as_,
                          float* __restrict__ ad_, int Nn, int C) {
  int node = blockIdx.x * (blockDim.x >> 5) + (threadIdx.x >> 5);
  int lane = threadIdx.x & 31;
  if (node >= Nn) return;
  const float* h = H + (size_t)node * C;
  float s = 0.f, d = 0.f;
  for (int c = lane; c < C; c += 32) {
    float hv = h[c];
    s += hv * vs[c];
    d += hv * vd[c];
  }
  for (int off = 16; off; off >>= 1) {
    s += __shfl_down(s, off, 32);
    d += __shfl_down(d, off, 32);
  }
  if (lane == 0) { as_[node] = s; ad_[node] = d; }
}

// ----------------------------------------------------- edge segment softmax

__global__ void edge_scores(const int* __restrict__ src, const int* __restrict__ dst,
                            const float* __restrict__ as_, const float* __restrict__ ad_,
                            float* __restrict__ eval, float* __restrict__ mbuf, int E) {
  for (int e = blockIdx.x * blockDim.x + threadIdx.x; e < E;
       e += gridDim.x * blockDim.x) {
    float v = as_[src[e]] + ad_[dst[e]];
    v = (v > 0.f) ? v : 0.2f * v;   // leaky_relu(0.2)
    eval[e] = v;
    atomicMaxF(&mbuf[dst[e]], v);
  }
}

__global__ void edge_exp(const int* __restrict__ dst, const float* __restrict__ eval,
                         const float* __restrict__ mbuf, float* __restrict__ ex,
                         float* __restrict__ sbuf, int E) {
  for (int e = blockIdx.x * blockDim.x + threadIdx.x; e < E;
       e += gridDim.x * blockDim.x) {
    float v = __expf(eval[e] - mbuf[dst[e]]);
    ex[e] = v;
    atomicAdd(&sbuf[dst[e]], v);
  }
}

__global__ void edge_norm(const int* __restrict__ dst, float* __restrict__ ex,
                          const float* __restrict__ sbuf, int E) {
  for (int e = blockIdx.x * blockDim.x + threadIdx.x; e < E;
       e += gridDim.x * blockDim.x)
    ex[e] = ex[e] / (sbuf[dst[e]] + 1e-16f);
}

// out[dst] += h[src] * alpha ; float4 vectorized scatter
__global__ void edge_agg(const int* __restrict__ src, const int* __restrict__ dst,
                         const float* __restrict__ alpha, const float* __restrict__ H,
                         float* __restrict__ out, int E, int C) {
  int cv = C >> 2;
  size_t total = (size_t)E * cv;
  for (size_t i = (size_t)blockIdx.x * blockDim.x + threadIdx.x; i < total;
       i += (size_t)gridDim.x * blockDim.x) {
    int e  = (int)(i / cv);
    int c4 = (int)(i % cv) * 4;
    float a = alpha[e];
    const float4 hv = *(const float4*)(H + (size_t)src[e] * C + c4);
    float* o = out + (size_t)dst[e] * C + c4;
    atomicAdd(o + 0, hv.x * a);
    atomicAdd(o + 1, hv.y * a);
    atomicAdd(o + 2, hv.z * a);
    atomicAdd(o + 3, hv.w * a);
  }
}

// -------------------------------------------------------------- activations

__global__ void act_elu(float* __restrict__ x, size_t n) {
  for (size_t i = (size_t)blockIdx.x * blockDim.x + threadIdx.x; i < n;
       i += (size_t)gridDim.x * blockDim.x) {
    float v = x[i];
    x[i] = (v > 0.f) ? v : (__expf(v) - 1.f);
  }
}

__global__ void act_sigmoid(float* __restrict__ x, size_t n) {
  for (size_t i = (size_t)blockIdx.x * blockDim.x + threadIdx.x; i < n;
       i += (size_t)gridDim.x * blockDim.x)
    x[i] = 1.f / (1.f + __expf(-x[i]));
}

__global__ void act_softplus_clip(float* __restrict__ x, size_t n) {
  for (size_t i = (size_t)blockIdx.x * blockDim.x + threadIdx.x; i < n;
       i += (size_t)gridDim.x * blockDim.x) {
    float v  = x[i];
    float sp = (v > 20.f) ? v : log1pf(__expf(v));
    x[i] = fminf(fmaxf(sp, 1e-4f), 1e4f);
  }
}

__global__ void act_expclip_scale(float* __restrict__ x, const float* __restrict__ sf,
                                  int C, size_t n) {
  for (size_t i = (size_t)blockIdx.x * blockDim.x + threadIdx.x; i < n;
       i += (size_t)gridDim.x * blockDim.x) {
    int row = (int)(i / C);
    x[i] = fminf(fmaxf(__expf(x[i]), 1e-5f), 1e6f) * sf[row];
  }
}

// ------------------------------------------------------------------- driver

static inline int gblocks(size_t n) {
  size_t b = (n + 255) / 256;
  if (b > (1u << 20)) b = (1u << 20);
  return (int)b;
}

extern "C" void kernel_launch(void* const* d_in, const int* in_sizes, int n_in,
                              void* d_out, int out_size, void* d_ws, size_t ws_size,
                              hipStream_t stream) {
  const float* feat  = (const float*)d_in[0];
  const int*   ei    = (const int*)d_in[1];
  const int*   src   = ei;
  const int*   dst   = ei + N_EDGES;
  const float* sf    = (const float*)d_in[2];
  const float* W1    = (const float*)d_in[3];
  const float* a1s   = (const float*)d_in[4];
  const float* a1d   = (const float*)d_in[5];
  const float* W2    = (const float*)d_in[6];
  const float* a2s   = (const float*)d_in[7];
  const float* a2d   = (const float*)d_in[8];
  const float* Wd    = (const float*)d_in[9];
  const float* dis_s = (const float*)d_in[10];
  const float* dis_d = (const float*)d_in[11];
  const float* Wm    = (const float*)d_in[12];
  const float* m_s   = (const float*)d_in[13];
  const float* m_d   = (const float*)d_in[14];
  const float* Wp    = (const float*)d_in[15];
  const float* p_s   = (const float*)d_in[16];
  const float* p_d   = (const float*)d_in[17];
  const float* gw1   = (const float*)d_in[18];
  const float* gb1   = (const float*)d_in[19];
  const float* gw2   = (const float*)d_in[20];
  const float* gb2   = (const float*)d_in[21];

  float* ws = (float*)d_ws;
  size_t o = 0;
  float* sumv   = ws + o; o += C_IN;
  float* mxv    = ws + o; o += C_IN;
  float* cscale = ws + o; o += C_IN;
  o += C_IN;                               // keep 16B alignment for big buffers
  float* asb    = ws + o; o += N_NODES;
  float* adb    = ws + o; o += N_NODES;
  float* mBuf   = ws + o; o += N_NODES;
  float* sBuf   = ws + o; o += N_NODES;
  float* eVal   = ws + o; o += N_EDGES;
  float* alphaA = ws + o; o += N_EDGES;    // conv1 alphas, reused by conv3 (tied attn)
  float* alphaB = ws + o; o += N_EDGES;    // working alphas
  float* H1     = ws + o; o += (size_t)N_NODES * C_HID;  // conv1 / conv3 pre-agg h
  float* XB     = ws + o; o += (size_t)N_NODES * C_HID;  // elu activations (x2 / h3)
  float* H2     = ws + o; o += (size_t)N_NODES * C_IN;   // conv2 / head pre-agg h

  float* out_mean = (float*)d_out;
  float* out_disp = out_mean + (size_t)N_NODES * C_IN;
  float* out_pi   = out_disp + (size_t)N_NODES * C_IN;
  float* out_h2   = out_pi   + (size_t)N_NODES * C_IN;

  auto edge_softmax = [&](const float* as_, const float* ad_, float* alpha) {
    fill_f32<<<64, 256, 0, stream>>>(mBuf, -INFINITY, (size_t)N_NODES);
    fill_f32<<<64, 256, 0, stream>>>(sBuf, 0.f, (size_t)N_NODES);
    edge_scores<<<3125, 256, 0, stream>>>(src, dst, as_, ad_, eVal, mBuf, N_EDGES);
    edge_exp<<<3125, 256, 0, stream>>>(dst, eVal, mBuf, alpha, sBuf, N_EDGES);
    edge_norm<<<3125, 256, 0, stream>>>(dst, alpha, sBuf, N_EDGES);
  };
  auto agg = [&](const float* H, const float* alpha, float* out, int C) {
    size_t nflt = (size_t)N_NODES * C;
    fill_f32<<<gblocks(nflt), 256, 0, stream>>>(out, 0.f, nflt);
    size_t total = (size_t)N_EDGES * (C >> 2);
    edge_agg<<<gblocks(total), 256, 0, stream>>>(src, dst, alpha, H, out, N_EDGES, C);
  };

  // ---- GCA channel gate --------------------------------------------------
  fill_f32<<<16, 256, 0, stream>>>(sumv, 0.f, (size_t)C_IN);
  fill_f32<<<16, 256, 0, stream>>>(mxv, -INFINITY, (size_t)C_IN);
  dim3 rg(C_IN / 256, 128);
  col_reduce<<<rg, 256, 0, stream>>>(feat, sumv, mxv, N_NODES, C_IN,
                                     (N_NODES + 127) / 128);
  gca_mlp<<<1, C_IN, 0, stream>>>(sumv, mxv, gw1, gb1, gw2, gb2, cscale, N_NODES);

  const int MT = (N_NODES + 63) / 64;

  // ---- conv1: h = (x*gate) @ W1 ------------------------------------------
  gemm_wmma<2, false, true><<<dim3(MT, C_HID / 64), 256, 0, stream>>>(
      feat, C_IN, cscale, W1, C_HID, H1, C_HID, N_NODES, C_IN);
  node_dots<<<(N_NODES + 7) / 8, 256, 0, stream>>>(H1, a1s, a1d, asb, adb,
                                                   N_NODES, C_HID);
  edge_softmax(asb, adb, alphaA);
  agg(H1, alphaA, XB, C_HID);
  act_elu<<<gblocks((size_t)N_NODES * C_HID), 256, 0, stream>>>(
      XB, (size_t)N_NODES * C_HID);

  // ---- conv2: h2 (returned raw) ------------------------------------------
  gemm_wmma<1, false, false><<<dim3(MT, C_OUT / 32), 256, 0, stream>>>(
      XB, C_HID, nullptr, W2, C_OUT, H2, C_OUT, N_NODES, C_HID);
  node_dots<<<(N_NODES + 7) / 8, 256, 0, stream>>>(H2, a2s, a2d, asb, adb,
                                                   N_NODES, C_OUT);
  edge_softmax(asb, adb, alphaB);
  agg(H2, alphaB, out_h2, C_OUT);

  // ---- conv3: W2^T, tied attention => alpha identical to conv1's --------
  gemm_wmma<2, true, false><<<dim3(MT, C_HID / 64), 256, 0, stream>>>(
      out_h2, C_OUT, nullptr, W2, C_OUT, H1, C_HID, N_NODES, C_OUT);
  agg(H1, alphaA, XB, C_HID);
  act_elu<<<gblocks((size_t)N_NODES * C_HID), 256, 0, stream>>>(
      XB, (size_t)N_NODES * C_HID);   // XB = h3

  // ---- output heads ------------------------------------------------------
  const size_t NO = (size_t)N_NODES * C_IN;
  // pi = sigmoid(gat(h3, Wp))
  gemm_wmma<2, false, false><<<dim3(MT, C_IN / 64), 256, 0, stream>>>(
      XB, C_HID, nullptr, Wp, C_IN, H2, C_IN, N_NODES, C_HID);
  node_dots<<<(N_NODES + 7) / 8, 256, 0, stream>>>(H2, p_s, p_d, asb, adb,
                                                   N_NODES, C_IN);
  edge_softmax(asb, adb, alphaB);
  agg(H2, alphaB, out_pi, C_IN);
  act_sigmoid<<<gblocks(NO), 256, 0, stream>>>(out_pi, NO);

  // disp = clip(softplus(gat(h3, Wd)))
  gemm_wmma<2, false, false><<<dim3(MT, C_IN / 64), 256, 0, stream>>>(
      XB, C_HID, nullptr, Wd, C_IN, H2, C_IN, N_NODES, C_HID);
  node_dots<<<(N_NODES + 7) / 8, 256, 0, stream>>>(H2, dis_s, dis_d, asb, adb,
                                                   N_NODES, C_IN);
  edge_softmax(asb, adb, alphaB);
  agg(H2, alphaB, out_disp, C_IN);
  act_softplus_clip<<<gblocks(NO), 256, 0, stream>>>(out_disp, NO);

  // mean = clip(exp(gat(h3, Wm))) * scale_factor
  gemm_wmma<2, false, false><<<dim3(MT, C_IN / 64), 256, 0, stream>>>(
      XB, C_HID, nullptr, Wm, C_IN, H2, C_IN, N_NODES, C_HID);
  node_dots<<<(N_NODES + 7) / 8, 256, 0, stream>>>(H2, m_s, m_d, asb, adb,
                                                   N_NODES, C_IN);
  edge_softmax(asb, adb, alphaB);
  agg(H2, alphaB, out_mean, C_IN);
  act_expclip_scale<<<gblocks(NO), 256, 0, stream>>>(out_mean, sf, C_IN, NO);
}
